// TILGraphClassifier_71330816852766
// MI455X (gfx1250) — compile-verified, matrix operand order checked by reading, not run on hardware
//
#include <hip/hip_runtime.h>
#include <stdint.h>

// ---------------------------------------------------------------------------
// 2-layer GAT for MI455X (gfx1250, wave32).
//  - Layer-1 GEMM (100000x256 @ 256x256) via v_wmma_f32_16x16x32_bf16:
//    * B (W1^T) fully register-resident (8 fragments / wave)
//    * A tile double-buffered in LDS, fp32->bf16 on the fly
//    * all 8 A fragments loaded as one DS clause -> 8 back-to-back WMMAs
//  - Edge softmax-aggregation via two atomic passes; h1 + accumulator are
//    ~L2-resident (192MB), so fp32 global atomics are the fast path.
// ---------------------------------------------------------------------------

typedef __attribute__((ext_vector_type(16))) __bf16 v16bf;
typedef __attribute__((ext_vector_type(8)))  float  v8f;

#define IN_CH     256
#define C1        256   // heads(2) * hid(128)
#define OUT_CH    4
#define NEG_SLOPE 0.2f
#define M_PER_BLOCK 10  // 16-row tiles per block in the WMMA GEMM
#define LDS_STRIDE 264  // 256 + 8 bf16 pad per row (bank-conflict avoidance)

// ------------------------- small helpers -----------------------------------

__device__ inline void atomicMaxF(float* addr, float v) {
    // Monotone int/uint trick: works for any mix of signs, init = -3e38.
    if (v >= 0.0f) atomicMax((int*)addr, __float_as_int(v));
    else           atomicMin((unsigned int*)addr, __float_as_uint(v));
}

// ------------------------- prep: W1^T in bf16 ------------------------------

__global__ void prep_w1t(const float* __restrict__ W1, __bf16* __restrict__ w1t) {
    int i = blockIdx.x * blockDim.x + threadIdx.x;   // 65536 elements
    int k = i >> 8;          // input-channel (row of W1)
    int n = i & 255;         // output-channel (col of W1)
    w1t[n * 256 + k] = (__bf16)W1[(size_t)k * 256 + n];
}

// ------------------------- layer-1 GEMM (WMMA) -----------------------------
// Block = 8 waves. Wave w owns output columns [nhalf*128 + w*16, +16).
__global__ __launch_bounds__(256)
void gemm1_kernel(const float* __restrict__ x, const __bf16* __restrict__ w1t,
                  float* __restrict__ h1, int n_nodes) {
    __shared__ __align__(16) __bf16 a_lds[2][16 * LDS_STRIDE];  // double buffer

    const int wave   = threadIdx.x >> 5;
    const int lane   = threadIdx.x & 31;
    const int mchunk = blockIdx.x >> 1;
    const int nhalf  = blockIdx.x & 1;
    const int n0     = nhalf * 128 + wave * 16;
    const int col    = n0 + (lane & 15);
    const int g      = lane >> 4;                      // lane-half (k lo/hi)

    union Frag { v16bf v; uint4 q[2]; };

    // Preload all 8 B fragments (whole K=256) for this wave's 16 columns.
    Frag bfrag[8];
#pragma unroll
    for (int kk = 0; kk < 8; ++kk) {
        const int kb = kk * 32 + g * 8;
        bfrag[kk].q[0] = *(const uint4*)(w1t + (size_t)col * 256 + kb);
        bfrag[kk].q[1] = *(const uint4*)(w1t + (size_t)col * 256 + kb + 16);
    }

    const int srow = threadIdx.x >> 4;        // staging: row 0..15
    const int scol = (threadIdx.x & 15) * 16; // staging: 16 floats each
    const int arow = lane & 15;

    float4 r0, r1, r2, r3;                    // in-flight global tile
    auto load_tile = [&](int m0) {
        const float4* xr = (const float4*)(x + (size_t)(m0 + srow) * IN_CH + scol);
        r0 = xr[0]; r1 = xr[1]; r2 = xr[2]; r3 = xr[3];
    };
    auto store_tile = [&](int buf) {
        union { __bf16 h[16]; uint4 q[2]; } u;
        const float f[16] = { r0.x, r0.y, r0.z, r0.w, r1.x, r1.y, r1.z, r1.w,
                              r2.x, r2.y, r2.z, r2.w, r3.x, r3.y, r3.z, r3.w };
#pragma unroll
        for (int j = 0; j < 16; ++j) u.h[j] = (__bf16)f[j];
        uint4* d = (uint4*)&a_lds[buf][srow * LDS_STRIDE + scol];
        d[0] = u.q[0]; d[1] = u.q[1];
    };

    const int m_base = mchunk * M_PER_BLOCK * 16;
    if (m_base >= n_nodes) return;

    load_tile(m_base);
    store_tile(0);
    __syncthreads();

    for (int it = 0; it < M_PER_BLOCK; ++it) {
        const int m0 = m_base + it * 16;
        if (m0 >= n_nodes) break;                       // uniform per block
        const int m_next = m0 + 16;
        const bool has_next = (it + 1 < M_PER_BLOCK) && (m_next < n_nodes);

        if (has_next) load_tile(m_next);  // global latency hides under WMMA

        // All 8 A fragments in one DS clause, then 8 chained WMMAs.
        Frag af[8];
#pragma unroll
        for (int kk = 0; kk < 8; ++kk) {
            const __bf16* ap = &a_lds[it & 1][arow * LDS_STRIDE + kk * 32 + g * 8];
            af[kk].q[0] = *(const uint4*)ap;
            af[kk].q[1] = *(const uint4*)(ap + 16);
        }
        v8f c = {};
#pragma unroll
        for (int kk = 0; kk < 8; ++kk)
            c = __builtin_amdgcn_wmma_f32_16x16x32_bf16(
                    false, af[kk].v, false, bfrag[kk].v, (short)0, c, false, false);

        // D layout: VGPR r -> row (m0 + g*8 + r), col = n0 + (lane&15)
        float* out = h1 + (size_t)(m0 + g * 8) * C1 + col;
#pragma unroll
        for (int r = 0; r < 8; ++r) out[(size_t)r * C1] = c[r];

        if (has_next) store_tile((it + 1) & 1);
        __syncthreads();
    }
}

// --------------- per-node attention logits (layer 1) -----------------------
// One wave per node: 4 simultaneous 128-dot-products via lane partials.
__global__ __launch_bounds__(256)
void node_logits1(const float* __restrict__ h1,
                  const float* __restrict__ a_src, const float* __restrict__ a_dst,
                  float* __restrict__ als, float* __restrict__ ald,
                  float* __restrict__ m1, int n_nodes) {
    const int node = blockIdx.x * 8 + (threadIdx.x >> 5);
    if (node >= n_nodes) return;
    const int lane = threadIdx.x & 31;
    float s0 = 0.f, s1 = 0.f, d0 = 0.f, d1 = 0.f;
#pragma unroll
    for (int i = 0; i < 8; ++i) {
        const int cidx = i * 32 + lane;           // head = cidx>>7 == (i>=4)
        const float v  = h1[(size_t)node * C1 + cidx];
        const float as = a_src[cidx];             // a_src1 flat (2*128)
        const float ad = a_dst[cidx];
        if (i < 4) { s0 += v * as; d0 += v * ad; }
        else       { s1 += v * as; d1 += v * ad; }
    }
#pragma unroll
    for (int off = 16; off > 0; off >>= 1) {
        s0 += __shfl_xor(s0, off, 32);  s1 += __shfl_xor(s1, off, 32);
        d0 += __shfl_xor(d0, off, 32);  d1 += __shfl_xor(d1, off, 32);
    }
    if (lane == 0) {
        als[node * 2 + 0] = s0;  als[node * 2 + 1] = s1;
        ald[node * 2 + 0] = d0;  ald[node * 2 + 1] = d1;
        m1[node * 2 + 0] = -3.0e38f; m1[node * 2 + 1] = -3.0e38f;
    }
}

// --------------- edge pass 1 (layer 1): segment max ------------------------
__global__ void edge1_max(const long long* __restrict__ ei, int E, int n_nodes,
                          const float* __restrict__ als, const float* __restrict__ ald,
                          float* __restrict__ m1) {
    const int i = blockIdx.x * blockDim.x + threadIdx.x;
    const int EA = E + n_nodes;
    if (i >= EA) return;
    int s, d;
    if (i < E) { s = (int)ei[i]; d = (int)ei[(size_t)E + i]; }
    else       { s = d = i - E; }                 // self-loop
#pragma unroll
    for (int h = 0; h < 2; ++h) {
        float e = als[s * 2 + h] + ald[d * 2 + h];
        e = (e > 0.f) ? e : NEG_SLOPE * e;
        atomicMaxF(&m1[d * 2 + h], e);
    }
}

// --------------- edge pass 2 (layer 1): exp + weighted aggregate -----------
// One wave per edge. Lane owns 8 *contiguous* channels -> 2x b128 gather,
// 8 fp32 atomic scatter-adds (L2-resident accumulator).
__global__ __launch_bounds__(256)
void edge1_agg(const long long* __restrict__ ei, int E, int n_nodes,
               const float* __restrict__ als, const float* __restrict__ ald,
               const float* __restrict__ m1, const float* __restrict__ h1,
               float* __restrict__ denom, float* __restrict__ outraw) {
    const int i = blockIdx.x * 8 + (threadIdx.x >> 5);
    const int lane = threadIdx.x & 31;
    const int EA = E + n_nodes;
    if (i >= EA) return;
    int s, d;
    if (i < E) { s = (int)ei[i]; d = (int)ei[(size_t)E + i]; }
    else       { s = d = i - E; }
    float w[2];
#pragma unroll
    for (int h = 0; h < 2; ++h) {
        float e = als[s * 2 + h] + ald[d * 2 + h];
        e = (e > 0.f) ? e : NEG_SLOPE * e;
        w[h] = __expf(e - m1[d * 2 + h]);
    }
    if (lane == 0) {
        atomicAdd(&denom[d * 2 + 0], w[0]);
        atomicAdd(&denom[d * 2 + 1], w[1]);
    }
    // lane owns channels [lane*8, lane*8+8); head = lane>>4
    const float4* hs = (const float4*)(h1 + (size_t)s * C1 + lane * 8);
    const float4 va = hs[0], vb = hs[1];
    const float wv = w[lane >> 4];
    float* od = outraw + (size_t)d * C1 + lane * 8;
    atomicAdd(&od[0], wv * va.x);  atomicAdd(&od[1], wv * va.y);
    atomicAdd(&od[2], wv * va.z);  atomicAdd(&od[3], wv * va.w);
    atomicAdd(&od[4], wv * vb.x);  atomicAdd(&od[5], wv * vb.y);
    atomicAdd(&od[6], wv * vb.z);  atomicAdd(&od[7], wv * vb.w);
}

// --------------- layer-1 finalize: normalize + bias + ReLU -----------------
__global__ void finalize1(const float* __restrict__ outraw, const float* __restrict__ denom,
                          const float* __restrict__ b1, float* __restrict__ h2, int n_nodes) {
    const int i = blockIdx.x * blockDim.x + threadIdx.x;
    if (i >= n_nodes * C1) return;
    const int node = i >> 8;
    const int cidx = i & 255;
    const float v = outraw[i] / denom[node * 2 + (cidx >> 7)] + b1[cidx];
    h2[i] = fmaxf(v, 0.f);
}

// --------------- layer 2: GEMV (256->4) + logits per node ------------------
__global__ __launch_bounds__(256)
void node_l2(const float* __restrict__ h2, const float* __restrict__ W2,
             const float* __restrict__ a_src2, const float* __restrict__ a_dst2,
             float* __restrict__ t4, float* __restrict__ al2s, float* __restrict__ al2d,
             float* __restrict__ m2, int n_nodes) {
    const int node = blockIdx.x * 8 + (threadIdx.x >> 5);
    if (node >= n_nodes) return;
    const int lane = threadIdx.x & 31;
    float acc[4] = {0.f, 0.f, 0.f, 0.f};
#pragma unroll
    for (int k = 0; k < 8; ++k) {
        const int cidx = k * 32 + lane;
        const float v = h2[(size_t)node * C1 + cidx];
#pragma unroll
        for (int o = 0; o < 4; ++o) acc[o] += v * W2[cidx * 4 + o];
    }
#pragma unroll
    for (int off = 16; off > 0; off >>= 1)
#pragma unroll
        for (int o = 0; o < 4; ++o) acc[o] += __shfl_xor(acc[o], off, 32);
    if (lane == 0) {
        float als_ = 0.f, ald_ = 0.f;
#pragma unroll
        for (int o = 0; o < 4; ++o) {
            t4[node * 4 + o] = acc[o];
            als_ += acc[o] * a_src2[o];
            ald_ += acc[o] * a_dst2[o];
        }
        al2s[node] = als_; al2d[node] = ald_; m2[node] = -3.0e38f;
    }
}

// --------------- edge passes (layer 2, 1 head, 4 channels) -----------------
__global__ void edge2_max(const long long* __restrict__ ei, int E, int n_nodes,
                          const float* __restrict__ als, const float* __restrict__ ald,
                          float* __restrict__ m2) {
    const int i = blockIdx.x * blockDim.x + threadIdx.x;
    const int EA = E + n_nodes;
    if (i >= EA) return;
    int s, d;
    if (i < E) { s = (int)ei[i]; d = (int)ei[(size_t)E + i]; }
    else       { s = d = i - E; }
    float e = als[s] + ald[d];
    e = (e > 0.f) ? e : NEG_SLOPE * e;
    atomicMaxF(&m2[d], e);
}

__global__ void edge2_agg(const long long* __restrict__ ei, int E, int n_nodes,
                          const float* __restrict__ als, const float* __restrict__ ald,
                          const float* __restrict__ m2, const float* __restrict__ t4,
                          float* __restrict__ denom, float* __restrict__ outraw) {
    const int i = blockIdx.x * blockDim.x + threadIdx.x;
    const int EA = E + n_nodes;
    if (i >= EA) return;
    int s, d;
    if (i < E) { s = (int)ei[i]; d = (int)ei[(size_t)E + i]; }
    else       { s = d = i - E; }
    float e = als[s] + ald[d];
    e = (e > 0.f) ? e : NEG_SLOPE * e;
    const float w = __expf(e - m2[d]);
    atomicAdd(&denom[d], w);
    const float4 tv = *(const float4*)(t4 + (size_t)s * 4);
    float* od = outraw + (size_t)d * 4;
    atomicAdd(&od[0], w * tv.x);  atomicAdd(&od[1], w * tv.y);
    atomicAdd(&od[2], w * tv.z);  atomicAdd(&od[3], w * tv.w);
}

// --------------- final: normalize + bias + log_softmax ---------------------
__global__ void final_out(const float* __restrict__ outraw, const float* __restrict__ denom,
                          const float* __restrict__ b2, float* __restrict__ out, int n_nodes) {
    const int n = blockIdx.x * blockDim.x + threadIdx.x;
    if (n >= n_nodes) return;
    const float dn = denom[n];
    float v[4], mx = -3.0e38f;
#pragma unroll
    for (int o = 0; o < 4; ++o) {
        v[o] = outraw[(size_t)n * 4 + o] / dn + b2[o];
        mx = fmaxf(mx, v[o]);
    }
    float sum = 0.f;
#pragma unroll
    for (int o = 0; o < 4; ++o) sum += __expf(v[o] - mx);
    const float lse = __logf(sum);
#pragma unroll
    for (int o = 0; o < 4; ++o) out[(size_t)n * 4 + o] = v[o] - mx - lse;
}

// ---------------------------------------------------------------------------

extern "C" void kernel_launch(void* const* d_in, const int* in_sizes, int n_in,
                              void* d_out, int out_size, void* d_ws, size_t ws_size,
                              hipStream_t stream) {
    const float*     x      = (const float*)d_in[0];
    const long long* ei     = (const long long*)d_in[1];   // int64 per reference
    const float*     W1     = (const float*)d_in[2];
    const float*     a_src1 = (const float*)d_in[3];
    const float*     a_dst1 = (const float*)d_in[4];
    const float*     b1     = (const float*)d_in[5];
    const float*     W2     = (const float*)d_in[6];
    const float*     a_src2 = (const float*)d_in[7];
    const float*     a_dst2 = (const float*)d_in[8];
    const float*     b2     = (const float*)d_in[9];

    const int N  = in_sizes[0] / IN_CH;   // 100000
    const int E  = in_sizes[1] / 2;       // 1600000
    const int EA = E + N;                 // + self loops

    // Workspace carve-out (~213 MB total).
    char* p = (char*)d_ws;
    auto alloc = [&](size_t bytes) -> char* {
        char* r = p; p += (bytes + 255) & ~size_t(255); return r;
    };
    __bf16* w1t    = (__bf16*)alloc((size_t)256 * 256 * 2);
    float* h1      = (float*)alloc((size_t)N * C1 * 4);   // reused as h2 later
    float* outraw1 = (float*)alloc((size_t)N * C1 * 4);
    float* als1    = (float*)alloc((size_t)N * 2 * 4);
    float* ald1    = (float*)alloc((size_t)N * 2 * 4);
    float* m1      = (float*)alloc((size_t)N * 2 * 4);
    float* den1    = (float*)alloc((size_t)N * 2 * 4);
    float* t4      = (float*)alloc((size_t)N * 4 * 4);
    float* al2s    = (float*)alloc((size_t)N * 4);
    float* al2d    = (float*)alloc((size_t)N * 4);
    float* m2      = (float*)alloc((size_t)N * 4);
    float* den2    = (float*)alloc((size_t)N * 4);
    float* outraw2 = (float*)alloc((size_t)N * 4 * 4);

    // Per-call re-init (graph-replay safe).
    hipMemsetAsync(outraw1, 0, (size_t)N * C1 * 4, stream);
    hipMemsetAsync(den1,    0, (size_t)N * 2 * 4,  stream);
    hipMemsetAsync(outraw2, 0, (size_t)N * 4 * 4,  stream);
    hipMemsetAsync(den2,    0, (size_t)N * 4,      stream);

    prep_w1t<<<256, 256, 0, stream>>>(W1, w1t);

    const int mchunks = (N + 16 * M_PER_BLOCK - 1) / (16 * M_PER_BLOCK);
    gemm1_kernel<<<mchunks * 2, 256, 0, stream>>>(x, w1t, h1, N);

    node_logits1<<<(N + 7) / 8, 256, 0, stream>>>(h1, a_src1, a_dst1, als1, ald1, m1, N);
    edge1_max<<<(EA + 255) / 256, 256, 0, stream>>>(ei, E, N, als1, ald1, m1);
    edge1_agg<<<(EA + 7) / 8, 256, 0, stream>>>(ei, E, N, als1, ald1, m1, h1, den1, outraw1);
    finalize1<<<(N * C1 + 255) / 256, 256, 0, stream>>>(outraw1, den1, b1, h1, N); // h2 := h1 buf

    node_l2<<<(N + 7) / 8, 256, 0, stream>>>(h1, W2, a_src2, a_dst2, t4, al2s, al2d, m2, N);
    edge2_max<<<(EA + 255) / 256, 256, 0, stream>>>(ei, E, N, al2s, al2d, m2);
    edge2_agg<<<(EA + 255) / 256, 256, 0, stream>>>(ei, E, N, al2s, al2d, m2, t4, den2, outraw2);
    final_out<<<(N + 255) / 256, 256, 0, stream>>>(outraw2, den2, b2, (float*)d_out, N);

    (void)n_in; (void)out_size; (void)ws_size;
}
